// SwinTransformer_16956530885167
// MI455X (gfx1250) — matrix operand order repaired and checked by reading, not compiled, not run on hardware
//
#include <hip/hip_runtime.h>
#include <math.h>

typedef __bf16 bf16;
typedef __attribute__((ext_vector_type(16))) __bf16 v16bf;
typedef __attribute__((ext_vector_type(8)))  __bf16 v8bf;
typedef __attribute__((ext_vector_type(8)))  float  v8f;
typedef int v4i __attribute__((vector_size(16)));   // matches async-LDS builtin param type

enum { F_BIAS=1, F_GELU=2, F_RESID=4, F_OUTF32=8, F_OUTBF16=16, F_QSCALE=32 };

#if __has_builtin(__builtin_amdgcn_global_load_async_to_lds_b128) && \
    __has_builtin(__builtin_amdgcn_s_wait_asynccnt)
#define HAVE_ASYNC_LDS 1
#else
#define HAVE_ASYNC_LDS 0
#endif

// ---------------------------------------------------------------------------
// WMMA helpers (gfx1250: V_WMMA_F32_16X16X32_BF16, wave32)
// ---------------------------------------------------------------------------
__device__ inline v8f wmma_bf16(v16bf a, v16bf b, v8f c) {
  return __builtin_amdgcn_wmma_f32_16x16x32_bf16(false, a, false, b, (short)0, c,
                                                 false, false);
}

// A-matrix 16x32 bf16 fragment. Row = lane&15. K mapping (contiguous, matches B):
// lanes 0-15 : e=0..15 -> K=e      (run [kb,kb+16))
// lanes 16-31: e=0..15 -> K=e+16   (run [kb+16,kb+32))
__device__ inline v16bf ldfragA(const bf16* row, int kb, int hi) {
  v16bf r;
  v8bf lo = *(const v8bf*)(row + kb + hi * 16);
  v8bf up = *(const v8bf*)(row + kb + hi * 16 + 8);
#pragma unroll
  for (int i = 0; i < 8; ++i) { r[i] = lo[i]; r[i + 8] = up[i]; }
  return r;
}

// B-matrix 32x16 bf16 fragment from row-major B^T ([N,K] weight layout).
// Column n = lane&15. lanes 0-15 hold K=kb..kb+15, lanes 16-31 hold K=kb+16..kb+31.
__device__ inline v16bf ldfragB(const bf16* row, int kb, int hi) {
  v16bf r;
  v8bf lo = *(const v8bf*)(row + kb + hi * 16);
  v8bf up = *(const v8bf*)(row + kb + hi * 16 + 8);
#pragma unroll
  for (int i = 0; i < 8; ++i) { r[i] = lo[i]; r[i + 8] = up[i]; }
  return r;
}

__device__ inline int kmapB(int e, int hi) { return e + hi * 16; }

// ---------------------------------------------------------------------------
// Small helper kernels
// ---------------------------------------------------------------------------
__global__ void k_cvt_bf16(const float* __restrict__ in, bf16* __restrict__ out, int n) {
  int i = blockIdx.x * 256 + threadIdx.x;
  if (i < n) out[i] = (bf16)in[i];
}

// patch embed: conv 4x4 stride 4, NCHW input -> tokens [B,3136,96] f32 (pre-LN)
__global__ void k_patch_embed(const float* __restrict__ x, const float* __restrict__ w,
                              const float* __restrict__ bias, float* __restrict__ out) {
  int id = blockIdx.x * 256 + threadIdx.x;
  const int TOT = 16 * 3136 * 96;
  if (id >= TOT) return;
  int co = id % 96;
  int t  = (id / 96) % 3136;
  int b  = id / (96 * 3136);
  int oh = t / 56, ow = t % 56;
  float acc = bias[co];
#pragma unroll
  for (int ci = 0; ci < 3; ++ci)
#pragma unroll
    for (int kh = 0; kh < 4; ++kh)
#pragma unroll
      for (int kw = 0; kw < 4; ++kw) {
        float xv = x[(((size_t)b * 3 + ci) * 224 + oh * 4 + kh) * 224 + ow * 4 + kw];
        float wv = w[((co * 3 + ci) * 4 + kh) * 4 + kw];
        acc += xv * wv;
      }
  out[((size_t)b * 3136 + t) * 96 + co] = acc;
}

// layernorm over last dim C, one wave per row; optional f32 and/or bf16 outputs
__global__ __launch_bounds__(32) void k_layernorm(const float* __restrict__ in,
                                                  const float* __restrict__ g,
                                                  const float* __restrict__ b,
                                                  float* __restrict__ outF,
                                                  bf16* __restrict__ outB, int C) {
  int row = blockIdx.x, lane = threadIdx.x;
  const float* p = in + (size_t)row * C;
  float s = 0.f, s2 = 0.f;
  for (int c = lane; c < C; c += 32) { float v = p[c]; s += v; s2 += v * v; }
#pragma unroll
  for (int m = 16; m >= 1; m >>= 1) { s += __shfl_xor(s, m, 32); s2 += __shfl_xor(s2, m, 32); }
  float mean = s / C;
  float var  = s2 / C - mean * mean;
  float inv  = rsqrtf(var + 1e-5f);
  for (int c = lane; c < C; c += 32) {
    float v = (p[c] - mean) * inv * g[c] + b[c];
    if (outF) outF[(size_t)row * C + c] = v;
    if (outB) outB[(size_t)row * C + c] = (bf16)v;
  }
}

// window partition with cyclic shift: LN'd [B,H,W,C] f32 -> [B*nW, 64, C] bf16 (rows 49..63=0)
__global__ void k_win_gather(const float* __restrict__ src, bf16* __restrict__ dst,
                             int H, int W, int C, int nWx, int shift, long total) {
  long id = (long)blockIdx.x * 256 + threadIdx.x;
  if (id >= total) return;
  int c   = (int)(id % C);
  int t   = (int)((id / C) % 64);
  int win = (int)(id / ((long)64 * C));
  if (t >= 49) { dst[id] = (bf16)0.0f; return; }
  int nWin = nWx * nWx;
  int b  = win / nWin;
  int wi = win % nWin;
  int wr = wi / nWx, wc = wi % nWx;
  int r = t / 7, cl = t % 7;
  int h = (wr * 7 + r + shift) % H;
  int w = (wc * 7 + cl + shift) % W;
  dst[id] = (bf16)src[(((size_t)b * H + h) * W + w) * C + c];
}

// window reverse (+unshift) and residual add: x = shortcut + scatter(proj)
__global__ void k_win_scatter_add(const float* __restrict__ proj,
                                  const float* __restrict__ shortcut,
                                  float* __restrict__ xout,
                                  int H, int W, int C, int nWx, int shift, long total) {
  long id = (long)blockIdx.x * 256 + threadIdx.x;
  if (id >= total) return;
  int c = (int)(id % C);
  int l = (int)((id / C) % ((long)H * W));
  int b = (int)(id / ((long)H * W * C));
  int h = l / W, w = l % W;
  int hp = (h - shift + H) % H;
  int wp = (w - shift + W) % W;
  int wr = hp / 7, r = hp % 7, wc = wp / 7, cl = wp % 7;
  int win = (b * nWx + wr) * nWx + wc;
  int t = r * 7 + cl;
  xout[id] = shortcut[id] + proj[((size_t)win * 64 + t) * C + c];
}

// patch merging gather (c*4 + i*2 + j channel order) + layernorm(4C) -> bf16 rows
__global__ __launch_bounds__(32) void k_merge_gather_ln(const float* __restrict__ x,
                                                        const float* __restrict__ g,
                                                        const float* __restrict__ b,
                                                        bf16* __restrict__ out,
                                                        int H, int W, int C) {
  int row = blockIdx.x, lane = threadIdx.x;
  int H2 = H / 2, W2 = W / 2;
  int b_ = row / (H2 * W2);
  int rr = row % (H2 * W2);
  int i2 = rr / W2, j2 = rr % W2;
  int C4 = 4 * C;
  float s = 0.f, s2 = 0.f;
  for (int k = lane; k < C4; k += 32) {
    int c = k >> 2, i = (k >> 1) & 1, j = k & 1;
    float v = x[(((size_t)b_ * H + (2 * i2 + i)) * W + (2 * j2 + j)) * C + c];
    s += v; s2 += v * v;
  }
#pragma unroll
  for (int m = 16; m >= 1; m >>= 1) { s += __shfl_xor(s, m, 32); s2 += __shfl_xor(s2, m, 32); }
  float mean = s / C4;
  float inv  = rsqrtf(s2 / C4 - mean * mean + 1e-5f);
  for (int k = lane; k < C4; k += 32) {
    int c = k >> 2, i = (k >> 1) & 1, j = k & 1;
    float v = x[(((size_t)b_ * H + (2 * i2 + i)) * W + (2 * j2 + j)) * C + c];
    out[(size_t)row * C4 + k] = (bf16)((v - mean) * inv * g[k] + b[k]);
  }
}

// final layernorm + [B,L,C] -> [B,C,H,W] transpose
__global__ __launch_bounds__(32) void k_final(const float* __restrict__ x,
                                              const float* __restrict__ g,
                                              const float* __restrict__ b,
                                              float* __restrict__ out, int C, int HW) {
  int row = blockIdx.x, lane = threadIdx.x;
  int bb = row / HW, l = row % HW;
  const float* p = x + (size_t)row * C;
  float s = 0.f, s2 = 0.f;
  for (int c = lane; c < C; c += 32) { float v = p[c]; s += v; s2 += v * v; }
#pragma unroll
  for (int m = 16; m >= 1; m >>= 1) { s += __shfl_xor(s, m, 32); s2 += __shfl_xor(s2, m, 32); }
  float mean = s / C;
  float inv  = rsqrtf(s2 / C - mean * mean + 1e-5f);
  for (int c = lane; c < C; c += 32) {
    float v = (p[c] - mean) * inv * g[c] + b[c];
    out[((size_t)bb * C + c) * HW + l] = v;
  }
}

// ---------------------------------------------------------------------------
// Generic bf16 WMMA GEMM: out[M,N] = A[M,K] * B[N,K]^T (+epilogues)
// block = 256 threads (8 waves), block tile 64x64, wave tile 16x32.
// K must be a multiple of 96 (true for all layers here). A/B tiles are staged
// into LDS with GLOBAL_LOAD_ASYNC_TO_LDS_B128 (ASYNCcnt) when available.
// M/N tails: staging may overshoot rows (callers guarantee buffer slack);
// garbage only feeds output rows/cols masked off in the epilogue.
// ---------------------------------------------------------------------------
__global__ __launch_bounds__(256) void k_gemm(const bf16* __restrict__ A,
                                              const bf16* __restrict__ Bw,
                                              const float* __restrict__ bias,
                                              float* __restrict__ outF,
                                              bf16* __restrict__ outB,
                                              const float* __restrict__ resid,
                                              int M, int N, int K, int flags,
                                              float qscale, int qcols) {
  __shared__ bf16 As[64 * 96];
  __shared__ bf16 Bs[64 * 96];
  int t = threadIdx.x;
  int lane = t & 31, wave = t >> 5;
  int hi = lane >> 4, lr = lane & 15;
  int mt = wave & 3, nh = wave >> 2;
  int mblk = blockIdx.x * 64;
  int nblk = blockIdx.y * 64;

  v8f acc0 = {0.f,0.f,0.f,0.f,0.f,0.f,0.f,0.f};
  v8f acc1 = {0.f,0.f,0.f,0.f,0.f,0.f,0.f,0.f};

  for (int k0 = 0; k0 < K; k0 += 96) {
    __syncthreads();  // previous chunk fully consumed
    // cooperative stage of A[64x96] and B[64x96]: 768 16B segments each,
    // 3 per thread per matrix.
#pragma unroll
    for (int i = 0; i < 3; ++i) {
      int seg = t + i * 256;               // 0..767
      int r = seg / 12, sc = (seg % 12) * 8;
      const bf16* ga = A  + (size_t)(mblk + r) * K + k0 + sc;
      const bf16* gb = Bw + (size_t)(nblk + r) * K + k0 + sc;
      bf16* la = As + r * 96 + sc;
      bf16* lb = Bs + r * 96 + sc;
      if (i == 0 && k0 + 96 < K) __builtin_prefetch(ga + 96, 0, 3);  // global_prefetch_b8
#if HAVE_ASYNC_LDS
      __builtin_amdgcn_global_load_async_to_lds_b128((v4i*)ga, (v4i*)la, 0, 0);
      __builtin_amdgcn_global_load_async_to_lds_b128((v4i*)gb, (v4i*)lb, 0, 0);
#else
      *(v8bf*)la = *(const v8bf*)ga;
      *(v8bf*)lb = *(const v8bf*)gb;
#endif
    }
#if HAVE_ASYNC_LDS
    __builtin_amdgcn_s_wait_asynccnt(0);
#endif
    __syncthreads();

    const bf16* ar  = As + (mt * 16 + lr) * 96;
    const bf16* br0 = Bs + (nh * 32 + lr) * 96;
    const bf16* br1 = Bs + (nh * 32 + 16 + lr) * 96;
#pragma unroll
    for (int kk = 0; kk < 96; kk += 32) {
      v16bf a  = ldfragA(ar,  kk, hi);
      v16bf b0 = ldfragB(br0, kk, hi);
      v16bf b1 = ldfragB(br1, kk, hi);
      acc0 = wmma_bf16(a, b0, acc0);
      acc1 = wmma_bf16(a, b1, acc1);
    }
  }

  int m0 = mblk + mt * 16;
  int n0 = nblk + nh * 32;
#pragma unroll
  for (int half = 0; half < 2; ++half) {
    int n = n0 + half * 16 + lr;
    if (n >= N) continue;
    float bv = (flags & F_BIAS) ? bias[n] : 0.f;
    float sc = ((flags & F_QSCALE) && n < qcols) ? qscale : 1.f;
#pragma unroll
    for (int v = 0; v < 8; ++v) {
      int m = m0 + v + 8 * hi;
      if (m >= M) continue;
      float val = (half ? acc1[v] : acc0[v]) + bv;
      val *= sc;
      if (flags & F_GELU) val = 0.5f * val * (1.f + erff(val * 0.70710678118654752f));
      if (flags & F_RESID) val += resid[(size_t)m * N + n];
      if (flags & F_OUTF32)  outF[(size_t)m * N + n] = val;
      if (flags & F_OUTBF16) outB[(size_t)m * N + n] = (bf16)val;
    }
  }
}

// ---------------------------------------------------------------------------
// Fused window multi-head self-attention: one wave per (window, head).
// qkv: [nWtot, 64, 3C] bf16 (q pre-scaled), out: [nWtot, 64, C] bf16.
// S = Q*K^T (+rel bias +shift mask), softmax in LDS, O = P*V. head_dim = 32.
// ---------------------------------------------------------------------------
__device__ inline int sw_region(int u, int dim) {
  return (u < dim - 7) ? 0 : ((u < dim - 3) ? 1 : 2);
}

__global__ __launch_bounds__(32) void k_attn(const bf16* __restrict__ qkv,
                                             const float* __restrict__ rel,
                                             bf16* __restrict__ out,
                                             int C, int heads, int H, int nWx, int shift) {
  __shared__ float Sl[64 * 72];
  __shared__ bf16  Pl[64 * 72];

  int win = blockIdx.x, hh = blockIdx.y;
  int lane = threadIdx.x, hi = lane >> 4, lr = lane & 15;
  int nWin = nWx * nWx;
  int wi = win % nWin;
  int wr = wi / nWx, wc = wi % nWx;
  int ld = 3 * C;
  const bf16* base = qkv + (size_t)win * 64 * ld;
  const bf16* Qp = base + hh * 32;
  const bf16* Kp = base + C + hh * 32;
  const bf16* Vp = base + 2 * C + hh * 32;

  // K fragments resident (B operand = rows of K, contiguous head_dim = 32)
  v16bf kf[4];
#pragma unroll
  for (int nt = 0; nt < 4; ++nt)
    kf[nt] = ldfragB(Kp + (size_t)(nt * 16 + lr) * ld, 0, hi);

  // S = Q*K^T tiles + bias + mask -> LDS
#pragma unroll
  for (int mt = 0; mt < 4; ++mt) {
    v16bf qf = ldfragA(Qp + (size_t)(mt * 16 + lr) * ld, 0, hi);
#pragma unroll
    for (int nt = 0; nt < 4; ++nt) {
      v8f s = {0.f,0.f,0.f,0.f,0.f,0.f,0.f,0.f};
      s = wmma_bf16(qf, kf[nt], s);
#pragma unroll
      for (int v = 0; v < 8; ++v) {
        int ntok = mt * 16 + v + 8 * hi;   // query token
        int mtok = nt * 16 + lr;           // key token
        float val;
        if (mtok >= 49 || ntok >= 49) {
          val = -1e30f;
        } else {
          int rn = ntok / 7, cn = ntok % 7, rm = mtok / 7, cm = mtok % 7;
          int ridx = (rn - rm + 6) * 13 + (cn - cm + 6);
          val = s[v] + rel[ridx * heads + hh];
          if (shift) {
            int cn_ = sw_region(wr * 7 + rn, H) * 3 + sw_region(wc * 7 + cn, H);
            int cm_ = sw_region(wr * 7 + rm, H) * 3 + sw_region(wc * 7 + cm, H);
            if (cn_ != cm_) val += -100.f;
          }
        }
        Sl[ntok * 72 + mtok] = val;
      }
    }
  }
  __syncthreads();

  // softmax: each lane handles rows lane and lane+32
#pragma unroll
  for (int rr = 0; rr < 2; ++rr) {
    int r = lane + rr * 32;
    float mx = -1e30f;
    for (int c = 0; c < 64; ++c) mx = fmaxf(mx, Sl[r * 72 + c]);
    float sum = 0.f;
    for (int c = 0; c < 64; ++c) { float e = __expf(Sl[r * 72 + c] - mx); Sl[r * 72 + c] = e; sum += e; }
    float inv = 1.f / sum;
    for (int c = 0; c < 64; ++c) Pl[r * 72 + c] = (bf16)(Sl[r * 72 + c] * inv);
  }
  __syncthreads();

  // V fragments (B operand = V[64,32]; element e -> V[k][n], strided gather)
  v16bf vf[2][2];
#pragma unroll
  for (int kt = 0; kt < 2; ++kt)
#pragma unroll
    for (int nt2 = 0; nt2 < 2; ++nt2) {
      v16bf f;
#pragma unroll
      for (int e = 0; e < 16; ++e)
        f[e] = Vp[(size_t)(kmapB(e, hi) + kt * 32) * ld + nt2 * 16 + lr];
      vf[kt][nt2] = f;
    }

  // O = P*V
#pragma unroll
  for (int mt = 0; mt < 4; ++mt) {
    v16bf pf0 = ldfragA(&Pl[(mt * 16 + lr) * 72], 0,  hi);
    v16bf pf1 = ldfragA(&Pl[(mt * 16 + lr) * 72], 32, hi);
    v8f o0 = {0.f,0.f,0.f,0.f,0.f,0.f,0.f,0.f};
    v8f o1 = {0.f,0.f,0.f,0.f,0.f,0.f,0.f,0.f};
    o0 = wmma_bf16(pf0, vf[0][0], o0); o0 = wmma_bf16(pf1, vf[1][0], o0);
    o1 = wmma_bf16(pf0, vf[0][1], o1); o1 = wmma_bf16(pf1, vf[1][1], o1);
#pragma unroll
    for (int v = 0; v < 8; ++v) {
      int tok = mt * 16 + v + 8 * hi;
      out[((size_t)win * 64 + tok) * C + hh * 32 + lr]      = (bf16)o0[v];
      out[((size_t)win * 64 + tok) * C + hh * 32 + 16 + lr] = (bf16)o1[v];
    }
  }
}

// ---------------------------------------------------------------------------
// Host orchestration
// ---------------------------------------------------------------------------
extern "C" void kernel_launch(void* const* d_in, const int* in_sizes, int n_in,
                              void* d_out, int out_size, void* d_ws, size_t ws_size,
                              hipStream_t stream) {
  (void)in_sizes; (void)n_in; (void)out_size; (void)ws_size;
  const int B = 16;
  static const int DEPTHS[4] = {2, 2, 6, 2};
  static const int HEADSA[4] = {3, 6, 12, 24};
  static const int BB[4]     = {0, 2, 4, 10};  // blocks before stage

  const float* xin = (const float*)d_in[0];
  auto PF = [&](int i) { return (const float*)d_in[i]; };
  // pytree (sorted-key) leaf order after x:
  auto MIDX = [&](int i, int leaf) { return 1 + i * 3 + leaf; };  // 0=ln_b,1=ln_g,2=red_w
  const int I_OUT_LN_B = 10, I_OUT_LN_G = 11, I_PATCH_B = 12, I_PATCH_W = 13,
            I_PE_LN_B = 14, I_PE_LN_G = 15;
  auto SIDX = [&](int s, int b, int leaf) { return 16 + (BB[s] + b) * 13 + leaf; };
  enum { L_FC1_B = 0, L_FC1_W, L_FC2_B, L_FC2_W, L_LN1_B, L_LN1_G, L_LN2_B, L_LN2_G,
         L_PROJ_B, L_PROJ_W, L_QKV_B, L_QKV_W, L_REL };

  // --- workspace carve (≈165 MB; GEMM staging may overshoot M/N rows, so every
  //     carve that feeds k_gemm keeps >=64*Kmax elements of slack) ---
  char* ws = (char*)d_ws;
  size_t off = 0;
  auto carve = [&](size_t bytes) -> char* {
    char* p = ws + off; off += (bytes + 255) & ~(size_t)255; return p;
  };
  const size_t SZ_X  = (size_t)B * 3136 * 96;    // residual stream f32
  const size_t SZ_AP = (size_t)B * 64 * 64 * 96; // max windows*64*C
  const size_t SZ_Q  = (size_t)B * 3136 * 384;   // max(qkv, mlp hidden) bf16
  float* X  = (float*)carve(SZ_X * 4);
  float* AP = (float*)carve(SZ_AP * 4);          // LN1 out, reused as proj out
  bf16* WIN = (bf16*)carve(SZ_AP * 2);           // windows / LN2 / merge staging
  bf16* QKV = (bf16*)carve(SZ_Q * 2);            // qkv / mlp hidden
  bf16* ATT = (bf16*)carve(SZ_AP * 2);
  bf16* WB  = (bf16*)carve((size_t)28 * 1024 * 1024 * 2);  // bf16 weights (+slack)

  // --- convert all GEMM weights to bf16 (every call; deterministic) ---
  size_t woff = 0;
  auto cvt = [&](const float* src, size_t n) -> bf16* {
    bf16* dst = WB + woff; woff += (n + 127) & ~(size_t)127;
    k_cvt_bf16<<<dim3((unsigned)((n + 255) / 256)), dim3(256), 0, stream>>>(src, dst, (int)n);
    return dst;
  };
  bf16 *qkvW[4][6], *projW[4][6], *fc1W[4][6], *fc2W[4][6], *redW[3];
  {
    int C = 96;
    for (int s = 0; s < 4; ++s) {
      for (int b = 0; b < DEPTHS[s]; ++b) {
        qkvW[s][b] = cvt(PF(SIDX(s, b, L_QKV_W)), (size_t)3 * C * C);
        projW[s][b] = cvt(PF(SIDX(s, b, L_PROJ_W)), (size_t)C * C);
        fc1W[s][b] = cvt(PF(SIDX(s, b, L_FC1_W)), (size_t)4 * C * C);
        fc2W[s][b] = cvt(PF(SIDX(s, b, L_FC2_W)), (size_t)4 * C * C);
      }
      if (s < 3) { redW[s] = cvt(PF(MIDX(s, 2)), (size_t)8 * C * C); C *= 2; }
    }
  }

  // --- patch embed + embed LN -> residual stream X ---
  {
    long tot = (long)B * 3136 * 96;
    k_patch_embed<<<dim3((unsigned)((tot + 255) / 256)), dim3(256), 0, stream>>>(
        xin, PF(I_PATCH_W), PF(I_PATCH_B), AP);
    k_layernorm<<<B * 3136, 32, 0, stream>>>(AP, PF(I_PE_LN_G), PF(I_PE_LN_B), X,
                                             (bf16*)nullptr, 96);
  }

  int Hc = 56, Cc = 96;
  const float QSC = 0.17677669529663687f;  // 1/sqrt(32)
  for (int s = 0; s < 4; ++s) {
    int heads = HEADSA[s];
    int L = Hc * Hc, rows = B * L;
    int nWx = Hc / 7, nWin = nWx * nWx, nWtot = B * nWin, Mwin = nWtot * 64;
    for (int bi = 0; bi < DEPTHS[s]; ++bi) {
      int shift = (bi & 1) ? 3 : 0;
      // LN1
      k_layernorm<<<rows, 32, 0, stream>>>(X, PF(SIDX(s, bi, L_LN1_G)),
                                           PF(SIDX(s, bi, L_LN1_B)), AP, (bf16*)nullptr, Cc);
      // window partition (+shift) -> bf16
      {
        long tot = (long)nWtot * 64 * Cc;
        k_win_gather<<<dim3((unsigned)((tot + 255) / 256)), dim3(256), 0, stream>>>(
            AP, WIN, Hc, Hc, Cc, nWx, shift, tot);
      }
      // QKV projection (q pre-scaled)
      k_gemm<<<dim3(Mwin / 64, (3 * Cc + 63) / 64), 256, 0, stream>>>(
          WIN, qkvW[s][bi], PF(SIDX(s, bi, L_QKV_B)), nullptr, QKV, nullptr,
          Mwin, 3 * Cc, Cc, F_BIAS | F_OUTBF16 | F_QSCALE, QSC, Cc);
      // fused window MSA
      k_attn<<<dim3(nWtot, heads), 32, 0, stream>>>(
          QKV, PF(SIDX(s, bi, L_REL)), ATT, Cc, heads, Hc, nWx, shift);
      // output projection
      k_gemm<<<dim3(Mwin / 64, (Cc + 63) / 64), 256, 0, stream>>>(
          ATT, projW[s][bi], PF(SIDX(s, bi, L_PROJ_B)), AP, nullptr, nullptr,
          Mwin, Cc, Cc, F_BIAS | F_OUTF32, 0.f, 0);
      // window reverse (+unshift) + residual add
      {
        long tot = (long)rows * Cc;
        k_win_scatter_add<<<dim3((unsigned)((tot + 255) / 256)), dim3(256), 0, stream>>>(
            AP, X, X, Hc, Hc, Cc, nWx, shift, tot);
      }
      // LN2 -> bf16, MLP (fc1+GELU, fc2+residual)
      k_layernorm<<<rows, 32, 0, stream>>>(X, PF(SIDX(s, bi, L_LN2_G)),
                                           PF(SIDX(s, bi, L_LN2_B)), nullptr, WIN, Cc);
      k_gemm<<<dim3((rows + 63) / 64, (4 * Cc + 63) / 64), 256, 0, stream>>>(
          WIN, fc1W[s][bi], PF(SIDX(s, bi, L_FC1_B)), nullptr, QKV, nullptr,
          rows, 4 * Cc, Cc, F_BIAS | F_GELU | F_OUTBF16, 0.f, 0);
      k_gemm<<<dim3((rows + 63) / 64, (Cc + 63) / 64), 256, 0, stream>>>(
          QKV, fc2W[s][bi], PF(SIDX(s, bi, L_FC2_B)), X, nullptr, X,
          rows, Cc, 4 * Cc, F_BIAS | F_OUTF32 | F_RESID, 0.f, 0);
    }
    if (s < 3) {
      int rows2 = B * (Hc / 2) * (Hc / 2);
      k_merge_gather_ln<<<rows2, 32, 0, stream>>>(X, PF(MIDX(s, 1)), PF(MIDX(s, 0)),
                                                  WIN, Hc, Hc, Cc);
      k_gemm<<<dim3((rows2 + 63) / 64, (2 * Cc + 63) / 64), 256, 0, stream>>>(
          WIN, redW[s], nullptr, X, nullptr, nullptr,
          rows2, 2 * Cc, 4 * Cc, F_OUTF32, 0.f, 0);
      Hc /= 2; Cc *= 2;
    }
  }

  // final LN + transpose to [B,768,7,7]
  k_final<<<B * 49, 32, 0, stream>>>(X, PF(I_OUT_LN_G), PF(I_OUT_LN_B),
                                     (float*)d_out, 768, 49);
}
